// L2XGSG_10909216931993
// MI455X (gfx1250) — matrix operand order repaired and checked by reading, not compiled, not run on hardware
//
#include <hip/hip_runtime.h>
#include <hip/hip_bf16.h>
#include <math.h>

typedef __attribute__((ext_vector_type(16))) _Float16 v16h;
typedef __attribute__((ext_vector_type(8)))  float    v8f;

#define HDIM 64
#define NGRAPH 512
#define TOPK_CHUNK 1024

// ---------------------------------------------------------------------------
// degree: deg[dst[e]] += 1   (edge-count per destination node, all edges)
// ---------------------------------------------------------------------------
__global__ void degree_kernel(const int* __restrict__ dst, float* __restrict__ deg, int E) {
    int e = blockIdx.x * blockDim.x + threadIdx.x;
    if (e >= E) return;
    atomicAdd(&deg[dst[e]], 1.0f);
}

// ---------------------------------------------------------------------------
// scatter-accumulate messages: agg[dst] += feat[src] * w   (D = 128 or 64)
// one thread per (edge, float4-lane)
// ---------------------------------------------------------------------------
__global__ void scatter_accum(const float* __restrict__ feat, const int* __restrict__ src,
                              const int* __restrict__ dst, const float* __restrict__ ew,
                              float* __restrict__ agg, int E, int D) {
    int t = blockIdx.x * blockDim.x + threadIdx.x;
    int perEdge = D >> 2;
    int e = t / perEdge;
    if (e >= E) return;
    int q = (t - e * perEdge) * 4;
    int s = src[e], d = dst[e];
    float w = ew ? ew[e] : 1.0f;
    const float4 v = *(const float4*)(feat + (size_t)s * D + q);
    float* o = agg + (size_t)d * D + q;
    atomicAdd(o + 0, v.x * w);
    atomicAdd(o + 1, v.y * w);
    atomicAdd(o + 2, v.z * w);
    atomicAdd(o + 3, v.w * w);
}

// ---------------------------------------------------------------------------
// Pack [Wl ; Wr] (2Ka x 64, f32) into per-lane WMMA B-fragments (f16) laid out
// exactly as v_wmma_f32_16x16x32_f16 expects:
//   fragment (kk, nt), lane l: col n = nt*16 + (l&15),
//   K = kk*32 + (l>=16 ? 8 : 0) + (j < 8 ? j : j + 8), j = 0..15
// One 32-byte contiguous chunk per lane -> single vector load in the GEMM.
// ---------------------------------------------------------------------------
__global__ void pack_bfrag(const float* __restrict__ Wl, const float* __restrict__ Wr,
                           _Float16* __restrict__ bfrag, int Ka) {
    int kk   = blockIdx.x >> 2;
    int nt   = blockIdx.x & 3;
    int lane = threadIdx.x;          // 32 threads
    int lr   = lane & 15;
    int hi   = lane >> 4;
    int n    = nt * 16 + lr;
    int kb0  = kk * 32 + hi * 8;
    _Float16* o = bfrag + ((size_t)blockIdx.x * 32 + lane) * 16;
#pragma unroll
    for (int j = 0; j < 16; ++j) {
        int kb = kb0 + (j < 8 ? j : j + 8);
        float v = (kb < Ka) ? Wl[(size_t)kb * HDIM + n]
                            : Wr[(size_t)(kb - Ka) * HDIM + n];
        o[j] = (_Float16)v;
    }
}

// ---------------------------------------------------------------------------
// Fused SAGE linear layer via WMMA (KA compile-time -> fully unrolled k-loop):
//   hout = relu( [agg/max(deg,1) | hin] (N x 2KA) @ Bfrag(2KA x 64) + bias )
// One wave = one 16-row tile, 4 column tiles of v_wmma_f32_16x16x32_f16.
// A lane layout per CDNA5 ISA 7.12.2; B fragments pre-packed by pack_bfrag.
// ---------------------------------------------------------------------------
template <int KA>
__global__ __launch_bounds__(128)
void sage_gemm_wmma(const float* __restrict__ agg, const float* __restrict__ deg,
                    const float* __restrict__ hin,
                    const _Float16* __restrict__ bfrag,
                    const float* __restrict__ bias,
                    float* __restrict__ hout, int N) {
    constexpr int H = HDIM;
    constexpr int KSTEPS = (2 * KA) >> 5;        // K total = 2*KA, 32 per WMMA
    int wave = threadIdx.x >> 5;
    int lane = threadIdx.x & 31;
    int lr   = lane & 15;
    int hi   = lane >> 4;                        // 0: lanes 0-15, 1: lanes 16-31
    int row0 = (blockIdx.x * 4 + wave) * 16;
    if (row0 >= N) return;                       // wave-uniform exit (EXEC stays full)
    int r = row0 + lr;
    if (r >= N) r = N - 1;                       // clamp; rows >= N never stored
    float invdeg = 1.0f / fmaxf(deg[r], 1.0f);

    v8f c[4];
#pragma unroll
    for (int nt = 0; nt < 4; ++nt) c[nt] = (v8f){0.f,0.f,0.f,0.f,0.f,0.f,0.f,0.f};

    const int koffA = hi * 8;                    // A-fragment K offset per half-wave

#pragma unroll
    for (int kk = 0; kk < KSTEPS; ++kk) {
        const int  k0    = kk << 5;
        const bool inAgg = (k0 < KA);
        const float* srcrow = inAgg ? (agg + (size_t)r * KA + k0)
                                    : (hin + (size_t)r * KA + (k0 - KA));
        const float scl = inAgg ? invdeg : 1.0f;

        // ---- A fragment: lane holds row r, K = {koffA..+7, koffA+16..+23} ----
        const float4 f0 = *(const float4*)(srcrow + koffA);
        const float4 f1 = *(const float4*)(srcrow + koffA + 4);
        const float4 f2 = *(const float4*)(srcrow + koffA + 16);
        const float4 f3 = *(const float4*)(srcrow + koffA + 20);
        v16h a;
        a[0]  = (_Float16)(f0.x * scl); a[1]  = (_Float16)(f0.y * scl);
        a[2]  = (_Float16)(f0.z * scl); a[3]  = (_Float16)(f0.w * scl);
        a[4]  = (_Float16)(f1.x * scl); a[5]  = (_Float16)(f1.y * scl);
        a[6]  = (_Float16)(f1.z * scl); a[7]  = (_Float16)(f1.w * scl);
        a[8]  = (_Float16)(f2.x * scl); a[9]  = (_Float16)(f2.y * scl);
        a[10] = (_Float16)(f2.z * scl); a[11] = (_Float16)(f2.w * scl);
        a[12] = (_Float16)(f3.x * scl); a[13] = (_Float16)(f3.y * scl);
        a[14] = (_Float16)(f3.z * scl); a[15] = (_Float16)(f3.w * scl);

        // ---- 4 column tiles: single 32B vector load per B fragment ----
#pragma unroll
        for (int nt = 0; nt < 4; ++nt) {
            const v16h b = *(const v16h*)(bfrag + (((size_t)kk * 4 + nt) * 32 + lane) * 16);
            c[nt] = __builtin_amdgcn_wmma_f32_16x16x32_f16(
                        false, a, false, b, (short)0, c[nt], false, false);
        }
    }

    // ---- epilogue: D VGPR v -> row row0 + v + hi*8, col nt*16 + lr ----
#pragma unroll
    for (int nt = 0; nt < 4; ++nt) {
        int col = nt * 16 + lr;
        float bv = bias[col];
#pragma unroll
        for (int v = 0; v < 8; ++v) {
            int row = row0 + v + hi * 8;
            if (row < N) {
                float val = c[nt][v] + bv;
                hout[(size_t)row * H + col] = fmaxf(val, 0.0f);
            }
        }
    }
}

// ---------------------------------------------------------------------------
// edge scores: s[e] = dot(h[src], h[dst]) over 64 dims; seg id + histogram
// ---------------------------------------------------------------------------
__global__ void edge_score_kernel(const float* __restrict__ h, const int* __restrict__ src,
                                  const int* __restrict__ dst, const int* __restrict__ batch,
                                  float* __restrict__ scores, int* __restrict__ segid,
                                  int* __restrict__ segcnt, int E) {
    int e = blockIdx.x * blockDim.x + threadIdx.x;
    if (e >= E) return;
    int s = src[e], d = dst[e];
    const float4* ps = (const float4*)(h + (size_t)s * HDIM);
    const float4* pd = (const float4*)(h + (size_t)d * HDIM);
    float acc = 0.0f;
#pragma unroll
    for (int i = 0; i < HDIM / 4; ++i) {
        float4 av = ps[i], bv = pd[i];
        acc += av.x * bv.x + av.y * bv.y + av.z * bv.z + av.w * bv.w;
    }
    scores[e] = acc;
    int g = batch[s];
    segid[e] = g;
    atomicAdd(&segcnt[g], 1);
}

// ---------------------------------------------------------------------------
// 512-wide inclusive scan -> exclusive offsets (single block)
// ---------------------------------------------------------------------------
__global__ void scan_kernel(const int* __restrict__ segcnt, int* __restrict__ segoff) {
    __shared__ int sh[NGRAPH];
    int t = threadIdx.x;
    sh[t] = segcnt[t];
    __syncthreads();
    for (int o = 1; o < NGRAPH; o <<= 1) {
        int v = (t >= o) ? sh[t - o] : 0;
        __syncthreads();
        sh[t] += v;
        __syncthreads();
    }
    segoff[t + 1] = sh[t];
    if (t == 0) segoff[0] = 0;
}

// ---------------------------------------------------------------------------
// bucket edges by segment
// ---------------------------------------------------------------------------
__global__ void bucket_kernel(const int* __restrict__ segid, const int* __restrict__ segoff,
                              int* __restrict__ cursor, int* __restrict__ bucket, int E) {
    int e = blockIdx.x * blockDim.x + threadIdx.x;
    if (e >= E) return;
    int g = segid[e];
    int p = atomicAdd(&cursor[g], 1);
    bucket[segoff[g] + p] = e;
}

// ---------------------------------------------------------------------------
// per-segment top-k (ratio = 0.5, k = ceil(m/2)) by exact rank counting.
// rank_i = #{j: s_j > s_i} + #{j: s_j == s_i && e_j < e_i}  (lexsort-stable)
// ---------------------------------------------------------------------------
__global__ void topk_kernel(const float* __restrict__ scores, const int* __restrict__ bucket,
                            const int* __restrict__ segoff,
                            float* __restrict__ sampled, float* __restrict__ ew) {
    int g = blockIdx.x;
    int beg = segoff[g], end = segoff[g + 1];
    int m = end - beg;
    int k = (m + 1) >> 1;                 // ceil(0.5 * m)
    __shared__ float s_sc[TOPK_CHUNK];
    __shared__ int   s_id[TOPK_CHUNK];

    int iters = (m + blockDim.x - 1) / blockDim.x;
    for (int ib = 0; ib < iters; ++ib) {
        int  i   = ib * blockDim.x + threadIdx.x;
        bool act = (i < m);
        int   ei = act ? bucket[beg + i] : -1;
        float si = act ? scores[ei] : 0.0f;
        int rank = 0;
        for (int c0 = 0; c0 < m; c0 += TOPK_CHUNK) {
            int cl = min(TOPK_CHUNK, m - c0);
            __syncthreads();
            for (int j = threadIdx.x; j < cl; j += blockDim.x) {
                int e2 = bucket[beg + c0 + j];
                s_id[j] = e2;
                s_sc[j] = scores[e2];
            }
            __syncthreads();
            if (act) {
                for (int j = 0; j < cl; ++j) {
                    float sj = s_sc[j];
                    rank += (sj > si) || (sj == si && s_id[j] < ei);
                }
            }
        }
        if (act) {
            int sel = (rank < k) ? 1 : 0;
            sampled[ei] = (float)sel;
            ew[ei] = sel ? si : 0.0f;
        }
    }
}

// ---------------------------------------------------------------------------
// graph mean-pool accumulation
// ---------------------------------------------------------------------------
__global__ void pool_kernel(const float* __restrict__ h, const int* __restrict__ batch,
                            float* __restrict__ pooled, float* __restrict__ gcnt, int N) {
    int t = blockIdx.x * blockDim.x + threadIdx.x;
    int n = t >> 4;
    if (n >= N) return;
    int q = (t & 15) * 4;
    int g = batch[n];
    const float4 v = *(const float4*)(h + (size_t)n * HDIM + q);
    float* o = pooled + (size_t)g * HDIM + q;
    atomicAdd(o + 0, v.x);
    atomicAdd(o + 1, v.y);
    atomicAdd(o + 2, v.z);
    atomicAdd(o + 3, v.w);
    if (q == 0) atomicAdd(&gcnt[g], 1.0f);
}

// ---------------------------------------------------------------------------
// MLP head + log_softmax, one block (64 threads) per graph
// ---------------------------------------------------------------------------
__global__ __launch_bounds__(64)
void head_kernel(const float* __restrict__ pooled, const float* __restrict__ gcnt,
                 const float* __restrict__ Wlin1, const float* __restrict__ blin1,
                 const float* __restrict__ Wlin2, const float* __restrict__ blin2,
                 float* __restrict__ out) {
    int g = blockIdx.x;
    int t = threadIdx.x;
    __shared__ float p[HDIM], z1[HDIM], z2[10];
    float inv = 1.0f / fmaxf(gcnt[g], 1.0f);
    p[t] = pooled[(size_t)g * HDIM + t] * inv;
    __syncthreads();
    float acc = blin1[t];
    for (int kk = 0; kk < HDIM; ++kk) acc += p[kk] * Wlin1[kk * HDIM + t];
    z1[t] = fmaxf(acc, 0.0f);
    __syncthreads();
    if (t < 10) {
        float a = blin2[t];
        for (int kk = 0; kk < HDIM; ++kk) a += z1[kk] * Wlin2[kk * 10 + t];
        z2[t] = a;
    }
    __syncthreads();
    if (t == 0) {
        float mx = z2[0];
        for (int i = 1; i < 10; ++i) mx = fmaxf(mx, z2[i]);
        float s = 0.0f;
        for (int i = 0; i < 10; ++i) s += __expf(z2[i] - mx);
        float l = logf(s);
        for (int i = 0; i < 10; ++i) out[g * 10 + i] = z2[i] - mx - l;
    }
}

// ---------------------------------------------------------------------------
extern "C" void kernel_launch(void* const* d_in, const int* in_sizes, int n_in,
                              void* d_out, int out_size, void* d_ws, size_t ws_size,
                              hipStream_t stream) {
    const int F = 128, H = HDIM, G = NGRAPH;
    const int N = in_sizes[0] / F;
    const int E = in_sizes[1] / 2;

    const float* x     = (const float*)d_in[0];
    const int*   ei    = (const int*)d_in[1];
    const int*   src   = ei;
    const int*   dst   = ei + E;
    const int*   batch = (const int*)d_in[2];
    const float* W1l = (const float*)d_in[3];  const float* b1l = (const float*)d_in[4];
    const float* W1r = (const float*)d_in[5];
    const float* W2l = (const float*)d_in[6];  const float* b2l = (const float*)d_in[7];
    const float* W2r = (const float*)d_in[8];
    const float* W3l = (const float*)d_in[9];  const float* b3l = (const float*)d_in[10];
    const float* W3r = (const float*)d_in[11];
    const float* Wlin1 = (const float*)d_in[12]; const float* blin1 = (const float*)d_in[13];
    const float* Wlin2 = (const float*)d_in[14]; const float* blin2 = (const float*)d_in[15];

    float* out_logits  = (float*)d_out;          // [G*10]
    float* out_sampled = (float*)d_out + G * 10; // [E]

    // workspace carve-up (256B aligned)
    char*  ws  = (char*)d_ws;
    size_t off = 0;
    auto alloc = [&](size_t bytes) {
        void* p = ws + off;
        off = (off + bytes + 255) & ~(size_t)255;
        return p;
    };
    float*     agg    = (float*)alloc((size_t)N * F * 4);   // reused for all 3 layers
    float*     deg    = (float*)alloc((size_t)N * 4);
    float*     h1     = (float*)alloc((size_t)N * H * 4);
    float*     h2     = (float*)alloc((size_t)N * H * 4);
    float*     h3     = (float*)alloc((size_t)N * H * 4);
    float*     scores = (float*)alloc((size_t)E * 4);
    float*     ew     = (float*)alloc((size_t)E * 4);
    int*       segid  = (int*)alloc((size_t)E * 4);
    int*       bucket = (int*)alloc((size_t)E * 4);
    int*       segcnt = (int*)alloc(G * 4);
    int*       segoff = (int*)alloc((G + 1) * 4);
    int*       cursor = (int*)alloc(G * 4);
    float*     pooled = (float*)alloc((size_t)G * H * 4);
    float*     gcnt   = (float*)alloc(G * 4);
    _Float16*  bfrag1 = (_Float16*)alloc((size_t)8 * 4 * 32 * 16 * 2);  // KSTEPS=8
    _Float16*  bfrag2 = (_Float16*)alloc((size_t)4 * 4 * 32 * 16 * 2);  // KSTEPS=4
    _Float16*  bfrag3 = (_Float16*)alloc((size_t)4 * 4 * 32 * 16 * 2);
    (void)ws_size; (void)n_in; (void)out_size;

    const int TB = 256;
    int gemm_blocks = (N + 63) / 64;

    // ---- zero accumulators (fresh every call: graph-replay safe) ----
    hipMemsetAsync(agg,    0, (size_t)N * F * 4, stream);
    hipMemsetAsync(deg,    0, (size_t)N * 4,     stream);
    hipMemsetAsync(segcnt, 0, G * 4,             stream);
    hipMemsetAsync(cursor, 0, G * 4,             stream);
    hipMemsetAsync(pooled, 0, (size_t)G * H * 4, stream);
    hipMemsetAsync(gcnt,   0, G * 4,             stream);

    // ---- pre-pack f16 WMMA B-fragments for the three SAGE layers ----
    pack_bfrag<<<8 * 4, 32, 0, stream>>>(W1l, W1r, bfrag1, F);
    pack_bfrag<<<4 * 4, 32, 0, stream>>>(W2l, W2r, bfrag2, H);
    pack_bfrag<<<4 * 4, 32, 0, stream>>>(W3l, W3r, bfrag3, H);

    // ---- layer 1 ----
    degree_kernel<<<(E + TB - 1) / TB, TB, 0, stream>>>(dst, deg, E);
    {
        long long tot = (long long)E * (F / 4);
        scatter_accum<<<(tot + TB - 1) / TB, TB, 0, stream>>>(x, src, dst, nullptr, agg, E, F);
    }
    sage_gemm_wmma<128><<<gemm_blocks, 128, 0, stream>>>(agg, deg, x, bfrag1, b1l, h1, N);

    // ---- edge scoring + per-graph top-k ----
    edge_score_kernel<<<(E + TB - 1) / TB, TB, 0, stream>>>(h1, src, dst, batch,
                                                            scores, segid, segcnt, E);
    scan_kernel<<<1, NGRAPH, 0, stream>>>(segcnt, segoff);
    bucket_kernel<<<(E + TB - 1) / TB, TB, 0, stream>>>(segid, segoff, cursor, bucket, E);
    topk_kernel<<<G, TB, 0, stream>>>(scores, bucket, segoff, out_sampled, ew);

    // ---- layer 2 (weighted messages) ----
    hipMemsetAsync(agg, 0, (size_t)N * H * 4, stream);
    {
        long long tot = (long long)E * (H / 4);
        scatter_accum<<<(tot + TB - 1) / TB, TB, 0, stream>>>(h1, src, dst, ew, agg, E, H);
    }
    sage_gemm_wmma<64><<<gemm_blocks, 128, 0, stream>>>(agg, deg, h1, bfrag2, b2l, h2, N);

    // ---- layer 3 ----
    hipMemsetAsync(agg, 0, (size_t)N * H * 4, stream);
    {
        long long tot = (long long)E * (H / 4);
        scatter_accum<<<(tot + TB - 1) / TB, TB, 0, stream>>>(h2, src, dst, ew, agg, E, H);
    }
    sage_gemm_wmma<64><<<gemm_blocks, 128, 0, stream>>>(agg, deg, h2, bfrag3, b3l, h3, N);

    // ---- pool + head ----
    {
        long long tot = (long long)N * (H / 4);
        pool_kernel<<<(tot + TB - 1) / TB, TB, 0, stream>>>(h3, batch, pooled, gcnt, N);
    }
    head_kernel<<<G, H, 0, stream>>>(pooled, gcnt, Wlin1, blin1, Wlin2, blin2, out_logits);
}
